// ProgressiveLeadLocalizationModelV48_87471303950791
// MI455X (gfx1250) — matrix operand order repaired and checked by reading, not compiled, not run on hardware
//
#include <hip/hip_runtime.h>
#include <hip/hip_bf16.h>
#include <hip/hip_fp16.h>

// ---------------------------------------------------------------------------
// MI455X (gfx1250) implementation notes:
//  * wave32; matrix ops via v_wmma_f32_16x16x32_f16.
//  * All inter-layer activations kept in f16; accumulate in f32.
//  * Conv = implicit GEMM: M=Cout, N=B*Ho*Wo, K=Cin*KH*KW. Block tile
//    64x64x32, 4 waves, each wave owns a 32x32 tile = 2x2 WMMA accumulators.
//  * Weight (A) tile staging: MODE 0 = global_load_async_to_lds_b128 +
//    s_wait_asynccnt; MODE 1 = Tensor Data Mover (tensor_load_to_lds with a
//    D# using LDS padding to reproduce the 96B row pitch) + s_wait_tensorcnt.
//  * Epilogue fuses BN-fold scale/bias (+conv bias), residual add, relu /
//    sigmoid, and channel-offset scatter for the fusion concat.
// ---------------------------------------------------------------------------

typedef _Float16 half_t;
typedef __attribute__((ext_vector_type(16))) _Float16 v16h;
typedef __attribute__((ext_vector_type(8)))  _Float16 v8h;
typedef __attribute__((ext_vector_type(8)))  float    v8f;
typedef __attribute__((ext_vector_type(4)))  unsigned int v4u;
typedef __attribute__((ext_vector_type(8)))  int v8i;
typedef __attribute__((ext_vector_type(4)))  int v4i;

#if defined(__has_builtin)
#  if __has_builtin(__builtin_amdgcn_tensor_load_to_lds)
#    define HAS_TDM 1
#  else
#    define HAS_TDM 0
#  endif
#else
#  define HAS_TDM 0
#endif

static inline int cdiv_i(int a, int b) { return (a + b - 1) / b; }
static inline int osz(int H, int K, int s, int p) { return (H + 2 * p - K) / s + 1; }

// ---------------------------------------------------------------------------
// Kernels
// ---------------------------------------------------------------------------

struct ConvP {
  int N, Cin, H, W;
  int Cout, KH, KW, stride, pad;
  int Ho, Wo;
  int yh_cs, yh_coff;   // f16 output channel stride / offset (for concat packing)
  int act;              // 0 none, 1 relu, 2 sigmoid
};

// Per-thread A-tile staging: async global->LDS b128 on the aligned fast path,
// scalar gather with zero-fill otherwise.
__device__ __forceinline__
void stage_A_tile(half_t (*Ash)[48], const half_t* __restrict__ Wt,
                  int m0, int k0, int Kdim, int Cout,
                  int arow, int acol, bool kvec) {
  const int mrow = m0 + arow;
  const bool mok = (mrow < Cout);
  const half_t* src = Wt + (size_t)mrow * Kdim + (k0 + acol);
  if (mok && kvec && (k0 + acol + 16 <= Kdim)) {
    unsigned ldsa = (unsigned)(size_t)&Ash[arow][acol];      // low 32 bits = LDS byte addr
    unsigned long long ga = (unsigned long long)(size_t)src;
    asm volatile("global_load_async_to_lds_b128 %0, %1, off"
                 :: "v"(ldsa), "v"(ga) : "memory");
    asm volatile("global_load_async_to_lds_b128 %0, %1, off"
                 :: "v"(ldsa + 16u), "v"(ga + 16ull) : "memory");
  } else {
    #pragma unroll
    for (int i = 0; i < 16; ++i) {
      int k = k0 + acol + i;
      Ash[arow][acol + i] = (mok && k < Kdim) ? src[i] : (half_t)0;
    }
  }
  if (mok) __builtin_prefetch(src + 32, 0, 1);   // global_prefetch_b8 on next K tile
}

// MODE 0: async-LDS weight staging. MODE 1: Tensor Data Mover weight staging.
template <int MODE>
__global__ __launch_bounds__(128)
void conv_implicit_gemm_wmma(const half_t* __restrict__ X,
                             const half_t* __restrict__ Wt,   // [Cout][Cin*KH*KW]
                             const float* __restrict__ scale, // per Cout
                             const float* __restrict__ bias,  // per Cout
                             const half_t* __restrict__ Res,  // optional residual (out layout)
                             half_t* __restrict__ Yh,         // optional f16 out
                             float* __restrict__ Yf,          // optional f32 out
                             ConvP p) {
  __shared__ half_t Ash[64][48];   // 48-half rows: 96B pitch, keeps b128 loads aligned
  __shared__ half_t Bsh[64][48];

  const int tid  = threadIdx.x;
  const int lane = tid & 31;
  const int wave = tid >> 5;
  const int lrow = lane & 15;
  const int hsel = lane >> 4;
  const int m0 = blockIdx.y * 64;
  const int n0 = blockIdx.x * 64;
  const int KHW  = p.KH * p.KW;
  const int Kdim = p.Cin * KHW;
  const int HoWo = p.Ho * p.Wo;
  const int Ngemm = p.N * HoWo;
  const int wm = (wave >> 1) * 32;
  const int wn = (wave & 1) * 32;
  const bool kvec = ((Kdim & 15) == 0);

  v8f acc[2][2] = {};

  const int arow = tid >> 1;            // 0..63
  const int acol = (tid & 1) * 16;      // 0 or 16
  const int nn = n0 + arow;
  int bb = 0, ho = 0, wo = 0;
  const bool nok = (nn < Ngemm);
  if (nok) {
    bb = nn / HoWo;
    int r = nn - bb * HoWo;
    ho = r / p.Wo;
    wo = r - ho * p.Wo;
  }
  const int hbase = ho * p.stride - p.pad;
  const int wbase = wo * p.stride - p.pad;

  for (int k0 = 0; k0 < Kdim; k0 += 32) {
    // ---- stage A tile (weights, row-major [Cout][K]) ----
    if constexpr (MODE == 1) {
#if HAS_TDM
      if (tid < 32) {
        // D# group 0: count=1, lds_addr, 57-bit global tile addr, type=2
        unsigned lds_base = (unsigned)(size_t)(&Ash[0][0]);
        unsigned long long ga =
            (unsigned long long)(size_t)(Wt + (size_t)m0 * Kdim + k0);
        unsigned td0 = (unsigned)(Kdim - k0);     // remaining K (zero-fill past end)
        unsigned td1 = (unsigned)(p.Cout - m0);   // remaining M
        v4u g0;
        g0[0] = 1u;
        g0[1] = lds_base;
        g0[2] = (unsigned)ga;
        g0[3] = (unsigned)((ga >> 32) & 0x01FFFFFFull) | (2u << 30);
        // D# group 1: data_size=2B, pad_enable, pad_interval=16 DW data,
        // pad_amount=8 DW -> 96B LDS row pitch; tile 32(K) x 64(M)
        v8i g1;
        g1[0] = (int)((1u << 16) | (1u << 20) | (3u << 22) | (7u << 25));
        g1[1] = (int)((td0 & 0xFFFFu) << 16);
        g1[2] = (int)(((td0 >> 16) & 0xFFFFu) | ((td1 & 0xFFFFu) << 16));
        g1[3] = (int)(((td1 >> 16) & 0xFFFFu) | (32u << 16));   // tile_dim0=32
        g1[4] = 64;                                             // tile_dim1=64
        g1[5] = (int)(unsigned)Kdim;                            // dim0 stride (elems)
        g1[6] = 0;
        g1[7] = 0;
        v4i gz = {0, 0, 0, 0};
#if __clang_major__ >= 23
        v8i gz8 = {0, 0, 0, 0, 0, 0, 0, 0};
        __builtin_amdgcn_tensor_load_to_lds(g0, g1, gz, gz, gz8, 0);
#else
        __builtin_amdgcn_tensor_load_to_lds(g0, g1, gz, gz, 0);
#endif
      }
#else
      stage_A_tile(Ash, Wt, m0, k0, Kdim, p.Cout, arow, acol, kvec);
#endif
    } else {
      stage_A_tile(Ash, Wt, m0, k0, Kdim, p.Cout, arow, acol, kvec);
    }

    // ---- stage B tile (im2col gather), stored n-major: Bsh[n][k] ----
    if (KHW == 1) {
      // 1x1 conv: k == cin, fixed spatial tap -> strided gather, no div/mod
      const size_t base = (((size_t)bb * p.Cin) * p.H + hbase) * p.W + wbase;
      const size_t cstride = (size_t)p.H * p.W;
      #pragma unroll
      for (int i = 0; i < 16; ++i) {
        int k = k0 + acol + i;
        Bsh[arow][acol + i] =
            (nok && k < Kdim) ? X[base + (size_t)k * cstride] : (half_t)0;
      }
    } else {
      #pragma unroll
      for (int i = 0; i < 16; ++i) {
        int k = k0 + acol + i;
        half_t v = (half_t)0;
        if (nok && k < Kdim) {
          int cin = k / KHW;
          int r = k - cin * KHW;
          int kh = r / p.KW;
          int kw = r - kh * p.KW;
          int hi = hbase + kh, wi = wbase + kw;
          if ((unsigned)hi < (unsigned)p.H && (unsigned)wi < (unsigned)p.W)
            v = X[(((size_t)bb * p.Cin + cin) * p.H + hi) * p.W + wi];
        }
        Bsh[arow][acol + i] = v;
      }
    }

    // async/TDM writes to LDS are not covered by DScnt: wait before barrier
    if constexpr (MODE == 1) {
#if HAS_TDM
      if (tid < 32) __builtin_amdgcn_s_wait_tensorcnt(0);
#else
      asm volatile("s_wait_asynccnt 0x0" ::: "memory");
#endif
    } else {
      asm volatile("s_wait_asynccnt 0x0" ::: "memory");
    }
    __syncthreads();

    // ---- WMMA: fragments per ISA 16-bit layouts ----
    v16h af[2], bf[2];
    #pragma unroll
    for (int i = 0; i < 2; ++i) {
      const half_t* ap = &Ash[wm + i * 16 + lrow][0];
      v8h lo = *(const v8h*)(ap + hsel * 8);        // K = hsel*8 + 0..7
      v8h hi = *(const v8h*)(ap + 16 + hsel * 8);   // K = 16 + hsel*8 + 0..7
      #pragma unroll
      for (int e = 0; e < 8; ++e) { af[i][e] = lo[e]; af[i][e + 8] = hi[e]; }
    }
    #pragma unroll
    for (int j = 0; j < 2; ++j) {
      const half_t* bp = &Bsh[wn + j * 16 + lrow][0];
      v8h lo = *(const v8h*)(bp + hsel * 8);        // K = hsel*8 + 0..7
      v8h hi = *(const v8h*)(bp + 16 + hsel * 8);   // K = 16 + hsel*8 + 0..7
      #pragma unroll
      for (int e = 0; e < 8; ++e) { bf[j][e] = lo[e]; bf[j][e + 8] = hi[e]; }
    }

    #pragma unroll
    for (int i = 0; i < 2; ++i)
      #pragma unroll
      for (int j = 0; j < 2; ++j)
        acc[i][j] = __builtin_amdgcn_wmma_f32_16x16x32_f16(
            false, af[i], false, bf[j], (short)0, acc[i][j], false, false);
    __syncthreads();
  }

  // ---- epilogue ----
  #pragma unroll
  for (int j = 0; j < 2; ++j) {
    int n = n0 + wn + j * 16 + lrow;
    if (n >= Ngemm) continue;
    int b2 = n / HoWo;
    int r2 = n - b2 * HoWo;
    int oh = r2 / p.Wo;
    int ow = r2 - oh * p.Wo;
    #pragma unroll
    for (int i = 0; i < 2; ++i) {
      #pragma unroll
      for (int rr = 0; rr < 8; ++rr) {
        int m = m0 + wm + i * 16 + hsel * 8 + rr;
        if (m >= p.Cout) continue;
        float vv = acc[i][j][rr] * scale[m] + bias[m];
        size_t ooff = (((size_t)b2 * p.Cout + m) * p.Ho + oh) * p.Wo + ow;
        if (Res) vv += (float)Res[ooff];
        if (p.act == 1) vv = fmaxf(vv, 0.f);
        else if (p.act == 2) vv = 1.f / (1.f + __expf(-vv));
        if (Yh) {
          size_t hoff = (((size_t)b2 * p.yh_cs + p.yh_coff + m) * p.Ho + oh) * p.Wo + ow;
          Yh[hoff] = (half_t)vv;
        }
        if (Yf) Yf[ooff] = vv;
      }
    }
  }
}

__global__ void cvt_f32_f16_kernel(const float* __restrict__ s, half_t* __restrict__ d, int n) {
  int i = blockIdx.x * blockDim.x + threadIdx.x;
  if (i < n) d[i] = (half_t)s[i];
}

// mode 0: BN fold; mode 1: BN fold + conv bias; mode 2: bias only (scale=1)
__global__ void make_affine_kernel(const float* g, const float* b, const float* m,
                                   const float* v, const float* cb,
                                   float* S, float* B, int C, int mode) {
  int i = blockIdx.x * blockDim.x + threadIdx.x;
  if (i >= C) return;
  if (mode == 2) { S[i] = 1.f; B[i] = cb ? cb[i] : 0.f; return; }
  float s = g[i] * rsqrtf(v[i] + 1e-5f);
  float bb = b[i] - m[i] * s;
  if (mode == 1) bb += cb[i] * s;
  S[i] = s; B[i] = bb;
}

__global__ void maxpool3s2_kernel(const half_t* __restrict__ x, half_t* __restrict__ y,
                                  int N, int C, int H, int W, int Ho, int Wo) {
  long long idx = (long long)blockIdx.x * blockDim.x + threadIdx.x;
  long long total = (long long)N * C * Ho * Wo;
  if (idx >= total) return;
  int ow = idx % Wo; long long t = idx / Wo;
  int oh = t % Ho; t /= Ho;
  int c = t % C; int n = (int)(t / C);
  const half_t* p = x + ((size_t)n * C + c) * H * W;
  float mv = -3.4e38f;
  for (int kh = 0; kh < 3; ++kh)
    for (int kw = 0; kw < 3; ++kw) {
      int hi = oh * 2 - 1 + kh, wi = ow * 2 - 1 + kw;
      if ((unsigned)hi < (unsigned)H && (unsigned)wi < (unsigned)W)
        mv = fmaxf(mv, (float)p[hi * W + wi]);
    }
  y[idx] = (half_t)mv;
}

__global__ void up2_add_kernel(half_t* __restrict__ y, const half_t* __restrict__ s,
                               int N, int C, int H, int W) {
  long long idx = (long long)blockIdx.x * blockDim.x + threadIdx.x;
  long long total = (long long)N * C * H * W;
  if (idx >= total) return;
  int w = idx % W; long long t = idx / W;
  int h = t % H; t /= H;
  int c = t % C; int n = (int)(t / C);
  int Hs = H / 2, Ws = W / 2;
  float v = (float)y[idx] + (float)s[(((size_t)n * C + c) * Hs + (h >> 1)) * Ws + (w >> 1)];
  y[idx] = (half_t)v;
}

__global__ void avgpool4_kernel(const half_t* __restrict__ x, half_t* __restrict__ y,
                                int N, int C, int H, int W) {
  int Ho = H / 4, Wo = W / 4;
  long long idx = (long long)blockIdx.x * blockDim.x + threadIdx.x;
  long long total = (long long)N * C * Ho * Wo;
  if (idx >= total) return;
  int ow = idx % Wo; long long t = idx / Wo;
  int oh = t % Ho; t /= Ho;
  int c = t % C; int n = (int)(t / C);
  const half_t* p = x + ((size_t)n * C + c) * H * W;
  float s = 0.f;
  for (int i = 0; i < 4; ++i)
    for (int j = 0; j < 4; ++j)
      s += (float)p[(oh * 4 + i) * W + ow * 4 + j];
  y[idx] = (half_t)(s * 0.0625f);
}

__global__ void pack_channels_kernel(const half_t* __restrict__ x, half_t* __restrict__ y,
                                     int N, int C, int H, int W, int yC, int coff) {
  long long idx = (long long)blockIdx.x * blockDim.x + threadIdx.x;
  long long total = (long long)N * C * H * W;
  if (idx >= total) return;
  int w = idx % W; long long t = idx / W;
  int h = t % H; t /= H;
  int c = t % C; int n = (int)(t / C);
  y[(((size_t)n * yC + coff + c) * H + h) * W + w] = x[idx];
}

__global__ void bilinear_up4_kernel(const half_t* __restrict__ x, half_t* __restrict__ y,
                                    int N, int Hs, int Ws, int yC, int coff) {
  int Ho = Hs * 4, Wo = Ws * 4;
  long long idx = (long long)blockIdx.x * blockDim.x + threadIdx.x;
  long long total = (long long)N * Ho * Wo;
  if (idx >= total) return;
  int w = idx % Wo; long long t = idx / Wo;
  int h = t % Ho; int n = (int)(t / Ho);
  float iy = fminf(fmaxf((h + 0.5f) * 0.25f - 0.5f, 0.f), (float)(Hs - 1));
  float ix = fminf(fmaxf((w + 0.5f) * 0.25f - 0.5f, 0.f), (float)(Ws - 1));
  int i0 = (int)iy, j0 = (int)ix;
  int i1 = min(i0 + 1, Hs - 1), j1 = min(j0 + 1, Ws - 1);
  float fy = iy - i0, fx = ix - j0;
  const half_t* p = x + (size_t)n * Hs * Ws;
  float v = (1.f - fy) * ((1.f - fx) * (float)p[i0 * Ws + j0] + fx * (float)p[i0 * Ws + j1]) +
            fy * ((1.f - fx) * (float)p[i1 * Ws + j0] + fx * (float)p[i1 * Ws + j1]);
  y[((size_t)n * yC + coff) * Ho * Wo + (size_t)h * Wo + w] = (half_t)v;
}

__global__ void maxpool2x1_kernel(const half_t* __restrict__ x, half_t* __restrict__ y,
                                  int N, int C, int H, int W) {
  int Ho = H / 2;
  long long idx = (long long)blockIdx.x * blockDim.x + threadIdx.x;
  long long total = (long long)N * C * Ho * W;
  if (idx >= total) return;
  int w = idx % W; long long t = idx / W;
  int oh = t % Ho; t /= Ho;
  int c = t % C; int n = (int)(t / C);
  const half_t* p = x + ((size_t)n * C + c) * H * W;
  float v = fmaxf((float)p[(2 * oh) * W + w], (float)p[(2 * oh + 1) * W + w]);
  y[idx] = (half_t)v;
}

__global__ __launch_bounds__(128)
void roi_centers_kernel(const float* __restrict__ base, float* __restrict__ ys,
                        float* __restrict__ ye, int L, int H, int W, int R) {
  int bl = blockIdx.x;
  const float* p = base + (size_t)bl * H * W;
  __shared__ float rnum[128], rden[128];
  float num = 0.f, den = 0.f;
  for (int h = threadIdx.x; h < H; h += blockDim.x) {
    float s = 0.f;
    for (int w = 0; w < W; ++w) s += p[h * W + w];
    float mean = s / (float)W;
    num += mean * (float)h;
    den += mean;
  }
  rnum[threadIdx.x] = num; rden[threadIdx.x] = den;
  __syncthreads();
  for (int st = 64; st > 0; st >>= 1) {
    if (threadIdx.x < st) {
      rnum[threadIdx.x] += rnum[threadIdx.x + st];
      rden[threadIdx.x] += rden[threadIdx.x + st];
    }
    __syncthreads();
  }
  if (threadIdx.x == 0) {
    float c = rnum[0] / (rden[0] + 1e-6f);
    float hv = (float)R * 0.5f;
    float yt = fminf(fmaxf(c - hv, 0.f), (float)(H - 1));
    float yb = fminf(fmaxf(c + hv, 0.f), (float)(H - 1));
    ys[bl] = floorf(yt);
    ye[bl] = floorf(yb);
  }
}

__global__ void roi_sample_kernel(const half_t* __restrict__ d2, const float* __restrict__ ys,
                                  const float* __restrict__ ye, half_t* __restrict__ rois,
                                  int B, int C, int H, int W, int L, int R) {
  long long idx = (long long)blockIdx.x * blockDim.x + threadIdx.x;
  long long total = (long long)B * L * C * R * W;
  if (idx >= total) return;
  int w = idx % W; long long t = idx / W;
  int r = t % R; t /= R;
  int c = t % C; t /= C;
  int bl = (int)t;
  int b = bl / L;
  float y0 = ys[bl], y1 = ye[bl];
  float yc = (y0 + y1) * 0.5f, ysc = (y1 - y0) / (float)H;
  float gy = -1.f + 2.f * (float)r / (float)(R - 1);
  float gym = gy * ysc + 2.f * yc / (float)H - 1.f;
  float iy = ((gym + 1.f) * (float)H - 1.f) * 0.5f;
  float iyf = floorf(iy);
  float fy = iy - iyf;
  int i0 = (int)iyf, i1 = i0 + 1;
  float wy0 = (1.f - fy) * ((i0 >= 0 && i0 < H) ? 1.f : 0.f);
  float wy1 = fy * ((i1 >= 0 && i1 < H) ? 1.f : 0.f);
  int i0c = min(max(i0, 0), H - 1), i1c = min(max(i1, 0), H - 1);
  float gx = -1.f + 2.f * (float)w / (float)(W - 1);
  float ixf = ((gx + 1.f) * (float)W - 1.f) * 0.5f;
  float ixw = floorf(ixf);
  float fx = ixf - ixw;
  int j0 = (int)ixw, j1 = j0 + 1;
  float wx0 = (1.f - fx) * ((j0 >= 0 && j0 < W) ? 1.f : 0.f);
  float wx1 = fx * ((j1 >= 0 && j1 < W) ? 1.f : 0.f);
  int j0c = min(max(j0, 0), W - 1), j1c = min(max(j1, 0), W - 1);
  const half_t* f = d2 + ((size_t)b * C + c) * H * W;
  float row0 = (float)f[i0c * W + j0c] * wy0 + (float)f[i1c * W + j0c] * wy1;
  float row1 = (float)f[i0c * W + j1c] * wy0 + (float)f[i1c * W + j1c] * wy1;
  rois[idx] = (half_t)(row0 * wx0 + row1 * wx1);
}

// GRU scan: gi layout (n, 3H, T); out f16 channel-major (n, cs, T) at coff.
__global__ __launch_bounds__(128)
void gru_scan_kernel(const float* __restrict__ gi, const float* __restrict__ wh,
                     const float* __restrict__ bh, half_t* __restrict__ out,
                     int T, int Hd, int reverse, int cs, int coff) {
  int n = blockIdx.x;
  int h = threadIdx.x;
  __shared__ float hs[128];
  hs[h] = 0.f;
  __syncthreads();
  const float* gbase = gi + (size_t)n * 3 * Hd * T;
  const float* w0 = wh + (size_t)h * Hd;
  const float* w1 = wh + (size_t)(Hd + h) * Hd;
  const float* w2 = wh + (size_t)(2 * Hd + h) * Hd;
  for (int stp = 0; stp < T; ++stp) {
    int t = reverse ? (T - 1 - stp) : stp;
    float ghr = bh[h], ghz = bh[Hd + h], ghn = bh[2 * Hd + h];
    for (int k = 0; k < Hd; ++k) {
      float hk = hs[k];
      ghr += w0[k] * hk; ghz += w1[k] * hk; ghn += w2[k] * hk;
    }
    float ir = gbase[(size_t)h * T + t];
    float iz = gbase[(size_t)(Hd + h) * T + t];
    float inn = gbase[(size_t)(2 * Hd + h) * T + t];
    float r = 1.f / (1.f + __expf(-(ir + ghr)));
    float z = 1.f / (1.f + __expf(-(iz + ghz)));
    float nn = tanhf(inn + r * ghn);
    float hn = (1.f - z) * nn + z * hs[h];
    __syncthreads();
    hs[h] = hn;
    out[((size_t)n * cs + coff + h) * T + t] = (half_t)hn;
    __syncthreads();
  }
}

// ---------------------------------------------------------------------------
// Host-side orchestration
// ---------------------------------------------------------------------------

struct WS {
  char* p; size_t off, cap;
  void* alloc(size_t n) { void* r = p + off; off += (n + 255) & ~(size_t)255; return r; }
};

struct BNp { const float *g, *b, *m, *v; };
struct Btl { BNp b1, b2, b3, dbn; const float *c1, *c2, *c3, *dw; int down; };
struct GruP { const float *bh, *bi, *wh, *wi; };
struct Aff { const float* s; const float* b; };

static half_t* to_f16(WS& ws, hipStream_t st, const float* src, size_t n) {
  half_t* d = (half_t*)ws.alloc(n * sizeof(half_t));
  cvt_f32_f16_kernel<<<cdiv_i((int)n, 256), 256, 0, st>>>(src, d, (int)n);
  return d;
}

static Aff affine_bn(WS& ws, hipStream_t st, BNp p, int C) {
  float* S = (float*)ws.alloc(C * 4);
  float* B = (float*)ws.alloc(C * 4);
  make_affine_kernel<<<cdiv_i(C, 256), 256, 0, st>>>(p.g, p.b, p.m, p.v, nullptr, S, B, C, 0);
  return {S, B};
}
static Aff affine_bn_bias(WS& ws, hipStream_t st, BNp p, const float* cb, int C) {
  float* S = (float*)ws.alloc(C * 4);
  float* B = (float*)ws.alloc(C * 4);
  make_affine_kernel<<<cdiv_i(C, 256), 256, 0, st>>>(p.g, p.b, p.m, p.v, cb, S, B, C, 1);
  return {S, B};
}
static Aff affine_bias(WS& ws, hipStream_t st, const float* cb, int C) {
  float* S = (float*)ws.alloc(C * 4);
  float* B = (float*)ws.alloc(C * 4);
  make_affine_kernel<<<cdiv_i(C, 256), 256, 0, st>>>(nullptr, nullptr, nullptr, nullptr, cb, S, B, C, 2);
  return {S, B};
}

static void launch_conv(hipStream_t st, const half_t* X, int N, int Cin, int H, int W,
                        const half_t* Wt16, int Cout, int KH, int KW, int stride, int pad,
                        Aff a, const half_t* Res, half_t* Yh, float* Yf, int act,
                        int yh_cs, int yh_coff) {
  ConvP p;
  p.N = N; p.Cin = Cin; p.H = H; p.W = W;
  p.Cout = Cout; p.KH = KH; p.KW = KW; p.stride = stride; p.pad = pad;
  p.Ho = osz(H, KH, stride, pad); p.Wo = osz(W, KW, stride, pad);
  p.yh_cs = (yh_cs > 0) ? yh_cs : Cout;
  p.yh_coff = yh_coff;
  p.act = act;
  int Ngemm = N * p.Ho * p.Wo;
  dim3 grid(cdiv_i(Ngemm, 64), cdiv_i(Cout, 64));
  bool use_tdm = false;
#if HAS_TDM
  use_tdm = (KH == 1 && KW == 1);   // bulk of GEMM FLOPs: TDM-staged weight tiles
#endif
  if (use_tdm)
    conv_implicit_gemm_wmma<1><<<grid, 128, 0, st>>>(X, Wt16, a.s, a.b, Res, Yh, Yf, p);
  else
    conv_implicit_gemm_wmma<0><<<grid, 128, 0, st>>>(X, Wt16, a.s, a.b, Res, Yh, Yf, p);
}

// allocates f16 output, converts weights, launches conv
static half_t* conv_f16(hipStream_t st, WS& ws, const half_t* X, int N, int Cin, int H, int W,
                        const float* w32, int Cout, int KH, int KW, int stride, int pad,
                        Aff a, const half_t* Res, int act, int& Ho, int& Wo) {
  Ho = osz(H, KH, stride, pad); Wo = osz(W, KW, stride, pad);
  half_t* Y = (half_t*)ws.alloc((size_t)N * Cout * Ho * Wo * sizeof(half_t));
  half_t* wt = to_f16(ws, st, w32, (size_t)Cout * Cin * KH * KW);
  launch_conv(st, X, N, Cin, H, W, wt, Cout, KH, KW, stride, pad, a, Res, Y, nullptr, act, Cout, 0);
  return Y;
}

static half_t* bottleneck(hipStream_t st, WS& ws, const Btl& t, const half_t* X,
                          int N, int& C, int& H, int& W, int planes, int stride) {
  int Cout = planes * 4;
  int Ho = osz(H, 3, stride, 1), Wo = osz(W, 3, stride, 1);
  half_t* out = (half_t*)ws.alloc((size_t)N * Cout * Ho * Wo * sizeof(half_t));
  size_t mk = ws.off;
  const half_t* idn = X;
  int dh, dw_;
  if (t.down)
    idn = conv_f16(st, ws, X, N, C, H, W, t.dw, Cout, 1, 1, stride, 0,
                   affine_bn(ws, st, t.dbn, Cout), nullptr, 0, dh, dw_);
  half_t* t1 = conv_f16(st, ws, X, N, C, H, W, t.c1, planes, 1, 1, 1, 0,
                        affine_bn(ws, st, t.b1, planes), nullptr, 1, dh, dw_);
  half_t* t2 = conv_f16(st, ws, t1, N, planes, H, W, t.c2, planes, 3, 3, stride, 1,
                        affine_bn(ws, st, t.b2, planes), nullptr, 1, dh, dw_);
  {
    half_t* wt = to_f16(ws, st, t.c3, (size_t)Cout * planes);
    Aff a = affine_bn(ws, st, t.b3, Cout);
    launch_conv(st, t2, N, planes, Ho, Wo, wt, Cout, 1, 1, 1, 0, a, idn, out, nullptr, 1, Cout, 0);
  }
  ws.off = mk;  // recycle temps (safe: single stream serializes reuse)
  C = Cout; H = Ho; W = Wo;
  return out;
}

extern "C" void kernel_launch(void* const* d_in, const int* in_sizes, int n_in,
                              void* d_out, int out_size, void* d_ws, size_t ws_size,
                              hipStream_t stream) {
  (void)in_sizes; (void)n_in; (void)out_size;
  hipStream_t st = stream;
  WS ws{(char*)d_ws, 0, ws_size};
  float* outF = (float*)d_out;

  // d_out slice offsets (coarse, text, wave, ocr, baselines, signals)
  const size_t OFF_COARSE = 0;
  const size_t OFF_TEXT   = 4096;
  const size_t OFF_WAVE   = 856064;
  const size_t OFF_OCR    = 1642496;
  const size_t OFF_BASE   = 1773568;
  const size_t OFF_SIG    = 2560000;

  // ---- walk the flattened parameter pytree (dict keys in sorted order) ----
  int ip = 1;  // d_in[0] == x
  auto nx = [&]() { return (const float*)d_in[ip++]; };
  auto rd_bn = [&]() { BNp p; p.g = nx(); p.b = nx(); p.m = nx(); p.v = nx(); return p; };
  auto rd_blk = [&](int down) {
    Btl t{}; t.b1 = rd_bn(); t.b2 = rd_bn(); t.b3 = rd_bn();
    t.c1 = nx(); t.c2 = nx(); t.c3 = nx();
    if (down) { t.dbn = rd_bn(); t.dw = nx(); }
    t.down = down; return t;
  };
  auto rd_gru = [&]() { GruP g; g.bh = nx(); g.bi = nx(); g.wh = nx(); g.wi = nx(); return g; };

  BNp bn1 = rd_bn();
  const float *coarse_w0 = nx(), *coarse_b0 = nx(), *coarse_w1 = nx(), *coarse_b1 = nx();
  const float* conv1_w = nx();
  // dec
  BNp db1 = rd_bn(), db2 = rd_bn(), db3 = rd_bn();
  const float *dc1w = nx(), *dc1b = nx(), *dc2w = nx(), *dc2b = nx(), *dc3w = nx(), *dc3b = nx();
  const float *h1w = nx(), *h1b = nx(), *h2w = nx(), *h2b = nx();
  GruP l0b = rd_gru(), l0f = rd_gru(), l1b = rd_gru(), l1f = rd_gru();
  // fusion
  BNp fb1 = rd_bn(), fb2 = rd_bn();
  const float *fc1w = nx(), *fc1b = nx(), *fc2w = nx(), *fc2b = nx(), *fc3w = nx(), *fc3b = nx();
  // laterals
  const float *lat1w = nx(), *lat1b = nx(), *lat2w = nx(), *lat2b = nx();
  const float *lat3w = nx(), *lat3b = nx(), *lat4w = nx(), *lat4b = nx();
  // backbone layers
  Btl L1[3], L2[4], L3[6], L4[3];
  L1[0] = rd_blk(1); for (int i = 1; i < 3; ++i) L1[i] = rd_blk(0);
  L2[0] = rd_blk(1); for (int i = 1; i < 4; ++i) L2[i] = rd_blk(0);
  L3[0] = rd_blk(1); for (int i = 1; i < 6; ++i) L3[i] = rd_blk(0);
  L4[0] = rd_blk(1); for (int i = 1; i < 3; ++i) L4[i] = rd_blk(0);
  const float *ocr_w0 = nx(), *ocr_b0 = nx(), *ocr_w1 = nx(), *ocr_b1 = nx();
  const float *smooth_w = nx(), *smooth_b = nx();
  const float *text_w0 = nx(), *text_b0 = nx(), *text_w1 = nx(), *text_b1 = nx();
  const float *wave_w0 = nx(), *wave_b0 = nx(), *wave_w1 = nx(), *wave_b1 = nx();

  // ---- stem ----
  const int N = 2;
  half_t* x16 = to_f16(ws, st, (const float*)d_in[0], (size_t)N * 3 * 512 * 1024);
  int C = 3, H = 512, W = 1024;
  int Ho, Wo;
  half_t* c1 = conv_f16(st, ws, x16, N, C, H, W, conv1_w, 64, 7, 7, 2, 3,
                        affine_bn(ws, st, bn1, 64), nullptr, 1, Ho, Wo);
  C = 64; H = Ho; W = Wo;                              // (2,64,256,512)
  int Hp = osz(H, 3, 2, 1), Wp = osz(W, 3, 2, 1);
  half_t* c1p = (half_t*)ws.alloc((size_t)N * C * Hp * Wp * sizeof(half_t));
  {
    long long tot = (long long)N * C * Hp * Wp;
    maxpool3s2_kernel<<<(int)((tot + 255) / 256), 256, 0, st>>>(c1, c1p, N, C, H, W, Hp, Wp);
  }
  H = Hp; W = Wp;                                       // (2,64,128,256)

  // ---- ResNet layers ----
  half_t* c2 = c1p;
  c2 = bottleneck(st, ws, L1[0], c2, N, C, H, W, 64, 1);
  for (int i = 1; i < 3; ++i) c2 = bottleneck(st, ws, L1[i], c2, N, C, H, W, 64, 1);
  int C2 = C, H2 = H, W2 = W;                           // (2,256,128,256)
  half_t* c3 = c2;
  c3 = bottleneck(st, ws, L2[0], c3, N, C, H, W, 128, 2);
  for (int i = 1; i < 4; ++i) c3 = bottleneck(st, ws, L2[i], c3, N, C, H, W, 128, 1);
  int C3 = C, H3 = H, W3 = W;                           // (2,512,64,128)
  half_t* c4 = c3;
  c4 = bottleneck(st, ws, L3[0], c4, N, C, H, W, 256, 2);
  for (int i = 1; i < 6; ++i) c4 = bottleneck(st, ws, L3[i], c4, N, C, H, W, 256, 1);
  int C4 = C, H4 = H, W4 = W;                           // (2,1024,32,64)
  half_t* c5 = c4;
  c5 = bottleneck(st, ws, L4[0], c5, N, C, H, W, 512, 2);
  for (int i = 1; i < 3; ++i) c5 = bottleneck(st, ws, L4[i], c5, N, C, H, W, 512, 1);
  int C5 = C, H5 = H, W5 = W;                           // (2,2048,16,32)

  // ---- FPN ----
  half_t* p5 = conv_f16(st, ws, c5, N, C5, H5, W5, lat4w, 256, 1, 1, 1, 0,
                        affine_bias(ws, st, lat4b, 256), nullptr, 0, Ho, Wo);
  half_t* p4 = conv_f16(st, ws, c4, N, C4, H4, W4, lat3w, 256, 1, 1, 1, 0,
                        affine_bias(ws, st, lat3b, 256), nullptr, 0, Ho, Wo);
  { long long tot = (long long)N * 256 * H4 * W4;
    up2_add_kernel<<<(int)((tot + 255) / 256), 256, 0, st>>>(p4, p5, N, 256, H4, W4); }
  half_t* p3 = conv_f16(st, ws, c3, N, C3, H3, W3, lat2w, 256, 1, 1, 1, 0,
                        affine_bias(ws, st, lat2b, 256), nullptr, 0, Ho, Wo);
  { long long tot = (long long)N * 256 * H3 * W3;
    up2_add_kernel<<<(int)((tot + 255) / 256), 256, 0, st>>>(p3, p4, N, 256, H3, W3); }
  half_t* p2 = conv_f16(st, ws, c2, N, C2, H2, W2, lat1w, 256, 1, 1, 1, 0,
                        affine_bias(ws, st, lat1b, 256), nullptr, 0, Ho, Wo);
  { long long tot = (long long)N * 256 * H2 * W2;
    up2_add_kernel<<<(int)((tot + 255) / 256), 256, 0, st>>>(p2, p3, N, 256, H2, W2); }
  half_t* d2 = conv_f16(st, ws, p2, N, 256, H2, W2, smooth_w, 256, 3, 3, 1, 1,
                        affine_bias(ws, st, smooth_b, 256), nullptr, 0, Ho, Wo);
  // d2: (2,256,128,256) f16

  // d4 = avgpool4(d2): (2,256,32,64)
  half_t* d4 = (half_t*)ws.alloc((size_t)N * 256 * (H2 / 4) * (W2 / 4) * sizeof(half_t));
  { long long tot = (long long)N * 256 * (H2 / 4) * (W2 / 4);
    avgpool4_kernel<<<(int)((tot + 255) / 256), 256, 0, st>>>(d2, d4, N, 256, H2, W2); }

  // fusion input (2,270,128,256) f16: [d2 | coarse_up | text]
  const int FINC = 270;
  half_t* fin = (half_t*)ws.alloc((size_t)N * FINC * H2 * W2 * sizeof(half_t));
  { long long tot = (long long)N * 256 * H2 * W2;
    pack_channels_kernel<<<(int)((tot + 255) / 256), 256, 0, st>>>(d2, fin, N, 256, H2, W2, FINC, 0); }

  // ---- heads ----
  size_t head_mk = ws.off;
  // coarse (on d4)
  half_t* hc = conv_f16(st, ws, d4, N, 256, H2 / 4, W2 / 4, coarse_w0, 128, 3, 3, 1, 1,
                        affine_bias(ws, st, coarse_b0, 128), nullptr, 1, Ho, Wo);
  half_t* coarse16 = (half_t*)ws.alloc((size_t)N * (H2 / 4) * (W2 / 4) * sizeof(half_t));
  {
    half_t* wt = to_f16(ws, st, coarse_w1, 128);
    launch_conv(st, hc, N, 128, H2 / 4, W2 / 4, wt, 1, 1, 1, 1, 0,
                affine_bias(ws, st, coarse_b1, 1), nullptr, coarse16, outF + OFF_COARSE, 2, 1, 0);
  }
  { long long tot = (long long)N * H2 * W2;
    bilinear_up4_kernel<<<(int)((tot + 255) / 256), 256, 0, st>>>(coarse16, fin, N, H2 / 4, W2 / 4, FINC, 256); }
  // text (on d2) -> d_out + fin channels 257..269
  half_t* ht = conv_f16(st, ws, d2, N, 256, H2, W2, text_w0, 128, 3, 3, 1, 1,
                        affine_bias(ws, st, text_b0, 128), nullptr, 1, Ho, Wo);
  {
    half_t* wt = to_f16(ws, st, text_w1, (size_t)13 * 128);
    launch_conv(st, ht, N, 128, H2, W2, wt, 13, 1, 1, 1, 0,
                affine_bias(ws, st, text_b1, 13), nullptr, fin, outF + OFF_TEXT, 2, FINC, 257);
  }
  // wave (on d2) -> d_out only (no sigmoid)
  half_t* hw = conv_f16(st, ws, d2, N, 256, H2, W2, wave_w0, 128, 3, 3, 1, 1,
                        affine_bias(ws, st, wave_b0, 128), nullptr, 1, Ho, Wo);
  {
    half_t* wt = to_f16(ws, st, wave_w1, (size_t)12 * 128);
    launch_conv(st, hw, N, 128, H2, W2, wt, 12, 1, 1, 1, 0,
                affine_bias(ws, st, wave_b1, 12), nullptr, nullptr, outF + OFF_WAVE, 0, 12, 0);
  }
  // ocr (on d2) -> d_out only (sigmoid)
  half_t* hoc = conv_f16(st, ws, d2, N, 256, H2, W2, ocr_w0, 64, 3, 3, 1, 1,
                         affine_bias(ws, st, ocr_b0, 64), nullptr, 1, Ho, Wo);
  {
    half_t* wt = to_f16(ws, st, ocr_w1, (size_t)2 * 64);
    launch_conv(st, hoc, N, 64, H2, W2, wt, 2, 1, 1, 1, 0,
                affine_bias(ws, st, ocr_b1, 2), nullptr, nullptr, outF + OFF_OCR, 2, 2, 0);
  }
  ws.off = head_mk;  // all consumers launched; recycle

  // ---- fusion -> baselines ----
  half_t* f1 = conv_f16(st, ws, fin, N, FINC, H2, W2, fc1w, 256, 3, 3, 1, 1,
                        affine_bn_bias(ws, st, fb1, fc1b, 256), nullptr, 1, Ho, Wo);
  half_t* f2 = conv_f16(st, ws, f1, N, 256, H2, W2, fc2w, 128, 3, 3, 1, 1,
                        affine_bn_bias(ws, st, fb2, fc2b, 128), nullptr, 1, Ho, Wo);
  {
    half_t* wt = to_f16(ws, st, fc3w, (size_t)12 * 128);
    launch_conv(st, f2, N, 128, H2, W2, wt, 12, 1, 1, 1, 0,
                affine_bias(ws, st, fc3b, 12), nullptr, nullptr, outF + OFF_BASE, 2, 12, 0);
  }

  // ---- ROI extraction ----
  const int Lc = 12, R = 32, NL = N * Lc;
  float* ysb = (float*)ws.alloc(NL * 4);
  float* yeb = (float*)ws.alloc(NL * 4);
  roi_centers_kernel<<<NL, 128, 0, st>>>(outF + OFF_BASE, ysb, yeb, Lc, H2, W2, R);
  half_t* rois = (half_t*)ws.alloc((size_t)NL * 256 * R * W2 * sizeof(half_t));
  { long long tot = (long long)NL * 256 * R * W2;
    roi_sample_kernel<<<(int)((tot + 255) / 256), 256, 0, st>>>(d2, ysb, yeb, rois, N, 256, H2, W2, Lc, R); }

  // ---- decoder ----
  size_t dec_mk = ws.off;
  half_t* dcx = conv_f16(st, ws, rois, NL, 256, R, W2, dc1w, 128, 3, 3, 1, 1,
                         affine_bn_bias(ws, st, db1, dc1b, 128), nullptr, 1, Ho, Wo);
  half_t* dp1 = (half_t*)ws.alloc((size_t)NL * 128 * (R / 2) * W2 * sizeof(half_t));
  { long long tot = (long long)NL * 128 * (R / 2) * W2;
    maxpool2x1_kernel<<<(int)((tot + 255) / 256), 256, 0, st>>>(dcx, dp1, NL, 128, R, W2); }
  half_t* dc2x = conv_f16(st, ws, dp1, NL, 128, R / 2, W2, dc2w, 64, 3, 3, 1, 1,
                          affine_bn_bias(ws, st, db2, dc2b, 64), nullptr, 1, Ho, Wo);
  half_t* dp2 = (half_t*)ws.alloc((size_t)NL * 64 * (R / 4) * W2 * sizeof(half_t));
  { long long tot = (long long)NL * 64 * (R / 4) * W2;
    maxpool2x1_kernel<<<(int)((tot + 255) / 256), 256, 0, st>>>(dc2x, dp2, NL, 64, R / 2, W2); }
  half_t* seq = conv_f16(st, ws, dp2, NL, 64, R / 4, W2, dc3w, 64, R / 4, 1, 1, 0,
                         affine_bn_bias(ws, st, db3, dc3b, 64), nullptr, 1, Ho, Wo);
  // seq: (24, 64, 1, 256)
  const int T = W2, Hd = 128;
  float* gi0f = (float*)ws.alloc((size_t)NL * 3 * Hd * T * 4);
  float* gi0b = (float*)ws.alloc((size_t)NL * 3 * Hd * T * 4);
  {
    half_t* wt = to_f16(ws, st, l0f.wi, (size_t)3 * Hd * 64);
    launch_conv(st, seq, NL, 64, 1, T, wt, 3 * Hd, 1, 1, 1, 0,
                affine_bias(ws, st, l0f.bi, 3 * Hd), nullptr, nullptr, gi0f, 0, 3 * Hd, 0);
    half_t* wtb = to_f16(ws, st, l0b.wi, (size_t)3 * Hd * 64);
    launch_conv(st, seq, NL, 64, 1, T, wtb, 3 * Hd, 1, 1, 1, 0,
                affine_bias(ws, st, l0b.bi, 3 * Hd), nullptr, nullptr, gi0b, 0, 3 * Hd, 0);
  }
  half_t* big0 = (half_t*)ws.alloc((size_t)NL * 2 * Hd * T * sizeof(half_t));
  gru_scan_kernel<<<NL, 128, 0, st>>>(gi0f, l0f.wh, l0f.bh, big0, T, Hd, 0, 2 * Hd, 0);
  gru_scan_kernel<<<NL, 128, 0, st>>>(gi0b, l0b.wh, l0b.bh, big0, T, Hd, 1, 2 * Hd, Hd);
  float* gi1f = (float*)ws.alloc((size_t)NL * 3 * Hd * T * 4);
  float* gi1b = (float*)ws.alloc((size_t)NL * 3 * Hd * T * 4);
  {
    half_t* wt = to_f16(ws, st, l1f.wi, (size_t)3 * Hd * 2 * Hd);
    launch_conv(st, big0, NL, 2 * Hd, 1, T, wt, 3 * Hd, 1, 1, 1, 0,
                affine_bias(ws, st, l1f.bi, 3 * Hd), nullptr, nullptr, gi1f, 0, 3 * Hd, 0);
    half_t* wtb = to_f16(ws, st, l1b.wi, (size_t)3 * Hd * 2 * Hd);
    launch_conv(st, big0, NL, 2 * Hd, 1, T, wtb, 3 * Hd, 1, 1, 1, 0,
                affine_bias(ws, st, l1b.bi, 3 * Hd), nullptr, nullptr, gi1b, 0, 3 * Hd, 0);
  }
  half_t* big1 = (half_t*)ws.alloc((size_t)NL * 2 * Hd * T * sizeof(half_t));
  gru_scan_kernel<<<NL, 128, 0, st>>>(gi1f, l1f.wh, l1f.bh, big1, T, Hd, 0, 2 * Hd, 0);
  gru_scan_kernel<<<NL, 128, 0, st>>>(gi1b, l1b.wh, l1b.bh, big1, T, Hd, 1, 2 * Hd, Hd);
  // head: relu(h @ w1.T + b1) @ w2.T + b2 -> signals
  half_t* hh = conv_f16(st, ws, big1, NL, 2 * Hd, 1, T, h1w, 64, 1, 1, 1, 0,
                        affine_bias(ws, st, h1b, 64), nullptr, 1, Ho, Wo);
  {
    half_t* wt = to_f16(ws, st, h2w, 64);
    launch_conv(st, hh, NL, 64, 1, T, wt, 1, 1, 1, 1, 0,
                affine_bias(ws, st, h2b, 1), nullptr, nullptr, outF + OFF_SIG, 0, 1, 0);
  }
  ws.off = dec_mk;  // done; everything consumed downstream already launched
}
